// NeuralLongTermMemory_75514114998738
// MI455X (gfx1250) — compile-verified, matrix operand order checked by reading, not run on hardware
//
#include <hip/hip_runtime.h>
#include <hip/hip_bf16.h>

// ---------------------------------------------------------------------------
// NeuralLongTermMemory for MI455X (gfx1250, wave32, WMMA, async-to-LDS).
// 8 large GEMMs ([8192x2048] x [2048x2048]^T) via v_wmma_f32_16x16x32_bf16.
// Block = 256 threads (8 waves) computes a 128x128 tile; K-slices of A and B
// are staged in LDS with double-buffered GLOBAL_LOAD_ASYNC_TO_LDS_B128.
// ---------------------------------------------------------------------------

typedef __bf16 bf16_t;
typedef __attribute__((ext_vector_type(16))) __bf16 v16bf;
typedef __attribute__((ext_vector_type(8)))  __bf16 v8bf;
typedef __attribute__((ext_vector_type(4)))  __bf16 v4bf;
typedef __attribute__((ext_vector_type(8)))  float  v8f;
typedef __attribute__((ext_vector_type(4)))  int    v4i_t;

#define NTOK 8192   // B*T
#define DDIM 2048
#define MEM_DECAY    0.01f
#define MEM_LR       0.1f
#define MEM_MOMENTUM 0.9f

#define MODE_SILU_STORE  0
#define MODE_SILU_COLSUM 1
#define MODE_SIG_COLSUM  2
#define MODE_STORE_BF16  3
#define MODE_STORE_F32   4

#define LDS_STRIDE 40   // 32 payload bf16 + 8 pad (16B) -> conflict-free b128 reads

static __device__ __forceinline__ bf16_t f32_to_bf16_rn(float f) {
    union { float f; unsigned u; } v; v.f = f;
    unsigned r = v.u + 0x7FFFu + ((v.u >> 16) & 1u);
    unsigned short h = (unsigned short)(r >> 16);
    bf16_t out;
    __builtin_memcpy(&out, &h, 2);
    return out;
}
static __device__ __forceinline__ float siluf(float x)     { return x / (1.0f + __expf(-x)); }
static __device__ __forceinline__ float sigmoidf_(float x) { return 1.0f / (1.0f + __expf(-x)); }

// ---- CDNA5 async global->LDS copy (ASYNCcnt), with safe fallback ----------
// Probe-learned signature: arg0 = AS1 (global) int4*, so arg1 = AS3 (LDS) int4*.
static __device__ __forceinline__ void async_copy_b128(const bf16_t* gsrc, bf16_t* lds_dst) {
#if defined(__has_builtin) && __has_builtin(__builtin_amdgcn_global_load_async_to_lds_b128)
    __builtin_amdgcn_global_load_async_to_lds_b128(
        (__attribute__((address_space(1))) v4i_t*)gsrc,
        (__attribute__((address_space(3))) v4i_t*)lds_dst, 0, 0);
#else
    *(v8bf*)lds_dst = *(const v8bf*)gsrc;   // sync fallback: load + ds_store
#endif
}
static __device__ __forceinline__ void wait_async_le4() {
#if defined(__has_builtin) && __has_builtin(__builtin_amdgcn_s_wait_asynccnt)
    __builtin_amdgcn_s_wait_asynccnt(4);
#else
    asm volatile("s_wait_asynccnt 0x4" ::: "memory");
#endif
}
static __device__ __forceinline__ void wait_async_le0() {
#if defined(__has_builtin) && __has_builtin(__builtin_amdgcn_s_wait_asynccnt)
    __builtin_amdgcn_s_wait_asynccnt(0);
#else
    asm volatile("s_wait_asynccnt 0x0" ::: "memory");
#endif
}

// ---------------------------------------------------------------------------
// Utility kernels
// ---------------------------------------------------------------------------
__global__ void k_f32_to_bf16(const float* __restrict__ in, bf16_t* __restrict__ out, size_t n4) {
    size_t i = (size_t)blockIdx.x * blockDim.x + threadIdx.x;
    size_t stride = (size_t)gridDim.x * blockDim.x;
    for (; i < n4; i += stride) {
        const float4 v = *(const float4*)(in + i * 4);
        v4bf o;
        o[0] = f32_to_bf16_rn(v.x); o[1] = f32_to_bf16_rn(v.y);
        o[2] = f32_to_bf16_rn(v.z); o[3] = f32_to_bf16_rn(v.w);
        *(v4bf*)(out + i * 4) = o;
    }
}

__global__ void k_zero_f32(float* __restrict__ p, int n) {
    int i = blockIdx.x * blockDim.x + threadIdx.x;
    if (i < n) p[i] = 0.0f;
}

// ---------------------------------------------------------------------------
// bf16 WMMA GEMM: C[m,n] = sum_k A[m,k]*Bw[n,k]  (A @ Bw^T)
// Block: 256 thr / 8 waves -> 128x128 tile. Wave: 32x64 (2x4 accumulators).
// K-slices (128x32 bf16, padded) double-buffered in LDS via async copies.
// ---------------------------------------------------------------------------
__global__ void __launch_bounds__(256)
k_gemm_bf16_wmma(const bf16_t* __restrict__ A, const bf16_t* __restrict__ Bw,
                 int M, int K, int Ncols, int mode,
                 const float* __restrict__ bias,
                 float* __restrict__ out_f32,
                 bf16_t* __restrict__ out_bf16,
                 float* __restrict__ colsum)
{
    __shared__ bf16_t As[2][128 * LDS_STRIDE];
    __shared__ bf16_t Bs[2][128 * LDS_STRIDE];
    __shared__ float  csum[128];

    const int tid  = threadIdx.x;
    const int lane = tid & 31;
    const int wave = tid >> 5;
    const int lm   = lane & 15;
    const int koff = (lane >> 4) << 3;        // 0 or 8
    const int waveM = wave & 3;               // 0..3  (32-row slabs)
    const int waveN = wave >> 2;              // 0..1  (64-col slabs)

    const int gridN  = Ncols >> 7;
    const int blockM = (blockIdx.x / gridN) << 7;
    const int blockN = (blockIdx.x % gridN) << 7;

    auto issue_copies = [&](int kpos, int buf) {
#pragma unroll
        for (int c = tid; c < 512; c += 256) {          // 2 iters: 4 async ops/wave
            const int row = c >> 2;
            const int seg = (c & 3) << 3;               // element offset, 16B chunks
            async_copy_b128(A  + (size_t)(blockM + row) * K + kpos + seg,
                            &As[buf][row * LDS_STRIDE + seg]);
            async_copy_b128(Bw + (size_t)(blockN + row) * K + kpos + seg,
                            &Bs[buf][row * LDS_STRIDE + seg]);
        }
    };

    v8f acc[2][4] = {};

    const int nk = K >> 5;
    issue_copies(0, 0);

    for (int kb = 0; kb < nk; ++kb) {
        const int cur = kb & 1;
        if (kb + 1 < nk) {
            issue_copies((kb + 1) << 5, 1 - cur);
            wait_async_le4();                 // in-order: older 4 (cur buffer) done
        } else {
            wait_async_le0();
        }
        __syncthreads();                      // cur slice visible to all waves

        const bf16_t* as = As[cur];
        const bf16_t* bs = Bs[cur];

        v16bf afrag[2];
#pragma unroll
        for (int mi = 0; mi < 2; ++mi) {
            const int arow = waveM * 32 + mi * 16 + lm;
            const v8bf alo = *(const v8bf*)(as + arow * LDS_STRIDE + koff);
            const v8bf ahi = *(const v8bf*)(as + arow * LDS_STRIDE + koff + 16);
#pragma unroll
            for (int i = 0; i < 8; ++i) { afrag[mi][i] = alo[i]; afrag[mi][i + 8] = ahi[i]; }
        }
#pragma unroll
        for (int t = 0; t < 4; ++t) {
            const int brow = waveN * 64 + t * 16 + lm;
            v16bf bfrag;
            const v8bf blo = *(const v8bf*)(bs + brow * LDS_STRIDE + koff);
            const v8bf bhi = *(const v8bf*)(bs + brow * LDS_STRIDE + koff + 16);
#pragma unroll
            for (int i = 0; i < 8; ++i) { bfrag[i] = blo[i]; bfrag[i + 8] = bhi[i]; }
            acc[0][t] = __builtin_amdgcn_wmma_f32_16x16x32_bf16(
                false, afrag[0], false, bfrag, (short)0, acc[0][t], false, false);
            acc[1][t] = __builtin_amdgcn_wmma_f32_16x16x32_bf16(
                false, afrag[1], false, bfrag, (short)0, acc[1][t], false, false);
        }
        __syncthreads();                      // reads of cur done before overwrite
    }

    // ------------------ epilogue ------------------
    // C layout: col = lane%16, row = r + 8*(lane>>4)
    const int rbase = (lane >> 4) << 3;
    if (mode == MODE_SILU_COLSUM || mode == MODE_SIG_COLSUM) {
        for (int i = tid; i < 128; i += 256) csum[i] = 0.f;
        __syncthreads();
#pragma unroll
        for (int t = 0; t < 4; ++t) {
            const int lc = waveN * 64 + t * 16 + lm;
            float s = 0.f;
            if (mode == MODE_SIG_COLSUM) {
                const float b = bias[blockN + lc];
#pragma unroll
                for (int mi = 0; mi < 2; ++mi)
#pragma unroll
                    for (int r = 0; r < 8; ++r) s += sigmoidf_(acc[mi][t][r] + b);
            } else {
#pragma unroll
                for (int mi = 0; mi < 2; ++mi)
#pragma unroll
                    for (int r = 0; r < 8; ++r) s += siluf(acc[mi][t][r]);
            }
            atomicAdd(&csum[lc], s);          // ds_add_f32
        }
        __syncthreads();
        for (int i = tid; i < 128; i += 256) atomicAdd(&colsum[blockN + i], csum[i]);
    } else {
#pragma unroll
        for (int mi = 0; mi < 2; ++mi) {
#pragma unroll
            for (int t = 0; t < 4; ++t) {
                const int col = blockN + waveN * 64 + t * 16 + lm;
                const size_t rowBase = (size_t)(blockM + waveM * 32 + mi * 16 + rbase);
                if (mode == MODE_SILU_STORE) {
#pragma unroll
                    for (int r = 0; r < 8; ++r)
                        out_f32[(rowBase + r) * Ncols + col] = siluf(acc[mi][t][r]);
                } else if (mode == MODE_STORE_BF16) {
#pragma unroll
                    for (int r = 0; r < 8; ++r)
                        out_bf16[(rowBase + r) * Ncols + col] = f32_to_bf16_rn(acc[mi][t][r]);
                } else {
#pragma unroll
                    for (int r = 0; r < 8; ++r)
                        out_f32[(rowBase + r) * Ncols + col] = acc[mi][t][r];
                }
            }
        }
    }
}

// ---------------------------------------------------------------------------
// Row l2-norm over 32 rows/block. mode 0: accumulate normalized rows into
// colsum (k_mean numerator, register partials -> few atomics). mode 1: write
// normalized rows as bf16 (q).
// ---------------------------------------------------------------------------
__global__ void __launch_bounds__(256)
k_rownorm32(const float* __restrict__ act, int D, int mode,
            float* __restrict__ colsum, bf16_t* __restrict__ outq)
{
    const int r0  = blockIdx.x << 5;
    const int tid = threadIdx.x;
    __shared__ float red[256];
    __shared__ float invs[32];

    {   // phase 1: squared norms, 8 threads per row, contiguous 256-elem chunks
        const int row = tid >> 3, part = tid & 7;
        const float* rp = act + (size_t)(r0 + row) * D + part * 256;
        float ss = 0.f;
#pragma unroll 8
        for (int i = 0; i < 256; i += 4) {
            const float4 v = *(const float4*)(rp + i);
            ss += v.x * v.x + v.y * v.y + v.z * v.z + v.w * v.w;
        }
        red[tid] = ss;
    }
    __syncthreads();
    if (tid < 32) {
        float ss = 0.f;
#pragma unroll
        for (int j = 0; j < 8; ++j) ss += red[tid * 8 + j];
        invs[tid] = 1.0f / fmaxf(__builtin_sqrtf(ss), 1e-12f);
    }
    __syncthreads();

    // phase 2: thread owns 8 contiguous columns across all 32 rows
    const int c0 = tid << 3;
    if (mode == 0) {
        float s[8] = {0.f, 0.f, 0.f, 0.f, 0.f, 0.f, 0.f, 0.f};
        for (int r = 0; r < 32; ++r) {
            const float inv = invs[r];
            const float4 v0 = *(const float4*)(act + (size_t)(r0 + r) * D + c0);
            const float4 v1 = *(const float4*)(act + (size_t)(r0 + r) * D + c0 + 4);
            s[0] += v0.x * inv; s[1] += v0.y * inv; s[2] += v0.z * inv; s[3] += v0.w * inv;
            s[4] += v1.x * inv; s[5] += v1.y * inv; s[6] += v1.z * inv; s[7] += v1.w * inv;
        }
#pragma unroll
        for (int j = 0; j < 8; ++j) atomicAdd(&colsum[c0 + j], s[j]);
    } else {
        for (int r = 0; r < 32; ++r) {
            const float inv = invs[r];
            const float4 v0 = *(const float4*)(act + (size_t)(r0 + r) * D + c0);
            const float4 v1 = *(const float4*)(act + (size_t)(r0 + r) * D + c0 + 4);
            v8bf o;
            o[0] = f32_to_bf16_rn(v0.x * inv); o[1] = f32_to_bf16_rn(v0.y * inv);
            o[2] = f32_to_bf16_rn(v0.z * inv); o[3] = f32_to_bf16_rn(v0.w * inv);
            o[4] = f32_to_bf16_rn(v1.x * inv); o[5] = f32_to_bf16_rn(v1.y * inv);
            o[6] = f32_to_bf16_rn(v1.z * inv); o[7] = f32_to_bf16_rn(v1.w * inv);
            *(v8bf*)(outq + (size_t)(r0 + r) * D + c0) = o;
        }
    }
}

// ---------------------------------------------------------------------------
__global__ void k_finalize_means(const float* __restrict__ k_sum, const float* __restrict__ v_sum,
                                 const float* __restrict__ d_sum, const float* __restrict__ lr_sum,
                                 const float* __restrict__ m_sum,
                                 float* __restrict__ k_mean, float* __restrict__ v_mean,
                                 float* __restrict__ alphav, float* __restrict__ thetav,
                                 float* __restrict__ etav, float invN, int D)
{
    int i = blockIdx.x * blockDim.x + threadIdx.x;
    if (i >= D) return;
    k_mean[i] = k_sum[i] * invN;
    v_mean[i] = v_sum[i] * invN;
    alphav[i] = d_sum[i]  * invN * MEM_DECAY;
    thetav[i] = lr_sum[i] * invN * MEM_LR;
    etav[i]   = m_sum[i]  * invN * MEM_MOMENTUM;
}

__global__ void __launch_bounds__(256)
k_matvec_err(const float* __restrict__ stateW, const float* __restrict__ k_mean,
             const float* __restrict__ v_mean, float* __restrict__ err, int D)
{
    const int e = blockIdx.x;
    const int tid = threadIdx.x;
    const float* __restrict__ rp = stateW + (size_t)e * D;
    float s = 0.f;
    for (int d = tid * 4; d < D; d += 1024) {
        const float4 a = *(const float4*)(rp + d);
        const float4 b = *(const float4*)(k_mean + d);
        s += a.x * b.x + a.y * b.y + a.z * b.z + a.w * b.w;
    }
    __shared__ float red[256];
    red[tid] = s;
    __syncthreads();
    for (int t = 128; t > 0; t >>= 1) {
        if (tid < t) red[tid] += red[tid + t];
        __syncthreads();
    }
    if (tid == 0) err[e] = red[0] - v_mean[e];
}

__global__ void k_build_wnew(const float* __restrict__ stateW, const float* __restrict__ stateM,
                             const float* __restrict__ alphav, const float* __restrict__ thetav,
                             const float* __restrict__ etav, const float* __restrict__ err,
                             const float* __restrict__ k_mean, bf16_t* __restrict__ wnew, int D)
{
    size_t i = (size_t)blockIdx.x * blockDim.x + threadIdx.x;
    const size_t total4 = (size_t)D * D / 4;
    const size_t stride = (size_t)gridDim.x * blockDim.x;
    const float twoOverD = 2.0f / (float)D;
    for (; i < total4; i += stride) {
        const size_t idx = i * 4;
        const int e = (int)(idx / D);
        const int d = (int)(idx % D);
        const float ce = twoOverD * err[e];
        const float eta = etav[e], th = thetav[e], oma = 1.0f - alphav[e];
        const float4 w = *(const float4*)(stateW + idx);
        const float4 m = *(const float4*)(stateM + idx);
        const float4 km = *(const float4*)(k_mean + d);
        v4bf o;
        o[0] = f32_to_bf16_rn(oma * w.x + eta * m.x - th * ce * km.x);
        o[1] = f32_to_bf16_rn(oma * w.y + eta * m.y - th * ce * km.y);
        o[2] = f32_to_bf16_rn(oma * w.z + eta * m.z - th * ce * km.z);
        o[3] = f32_to_bf16_rn(oma * w.w + eta * m.w - th * ce * km.w);
        *(v4bf*)(wnew + idx) = o;
    }
}

// ---------------------------------------------------------------------------
// launch
// ---------------------------------------------------------------------------
extern "C" void kernel_launch(void* const* d_in, const int* in_sizes, int n_in,
                              void* d_out, int out_size, void* d_ws, size_t ws_size,
                              hipStream_t stream) {
    (void)in_sizes; (void)n_in; (void)out_size; (void)ws_size;

    const size_t N = NTOK, D = DDIM, DD = D * D;

    const float* x_f32   = (const float*)d_in[0];
    const float* state_W = (const float*)d_in[1];
    const float* state_M = (const float*)d_in[2];
    const float* bd      = (const float*)d_in[8];
    const float* blr     = (const float*)d_in[10];
    const float* bm      = (const float*)d_in[12];

    unsigned char* ws = (unsigned char*)d_ws;
    size_t off = 0;
    bf16_t* x_bf = (bf16_t*)(ws + off); off += N * D * sizeof(bf16_t);     // 32 MB
    bf16_t* w_bf = (bf16_t*)(ws + off); off += 8 * DD * sizeof(bf16_t);    // 64 MB
    size_t act_off = off;
    float*  act  = (float*)(ws + off);  off += N * D * sizeof(float);      // 64 MB
    bf16_t* q_bf = (bf16_t*)(ws + off); off += N * D * sizeof(bf16_t);     // 32 MB
    bf16_t* retr_bf = (bf16_t*)(ws + act_off);                             // aliases act
    float* k_sum  = (float*)(ws + off); off += D * sizeof(float);
    float* v_sum  = (float*)(ws + off); off += D * sizeof(float);
    float* d_sum  = (float*)(ws + off); off += D * sizeof(float);
    float* lr_sum = (float*)(ws + off); off += D * sizeof(float);
    float* m_sum  = (float*)(ws + off); off += D * sizeof(float);
    float* k_mean = (float*)(ws + off); off += D * sizeof(float);
    float* v_mean = (float*)(ws + off); off += D * sizeof(float);
    float* alphav = (float*)(ws + off); off += D * sizeof(float);
    float* thetav = (float*)(ws + off); off += D * sizeof(float);
    float* etav   = (float*)(ws + off); off += D * sizeof(float);
    float* errv   = (float*)(ws + off); off += D * sizeof(float);

    k_zero_f32<<<(int)((5 * D + 255) / 256), 256, 0, stream>>>(k_sum, (int)(5 * D));

    k_f32_to_bf16<<<2048, 256, 0, stream>>>(x_f32, x_bf, N * D / 4);
    const int widx[7] = {3, 4, 5, 7, 9, 11, 6};   // Wk,Wv,Wq,Wd,Wlr,Wm,Wout
    for (int i = 0; i < 7; ++i)
        k_f32_to_bf16<<<1024, 256, 0, stream>>>((const float*)d_in[widx[i]],
                                                w_bf + (size_t)i * DD, DD / 4);

    const int GB = (int)((N / 128) * (D / 128));  // 1024 blocks

    // k path
    k_gemm_bf16_wmma<<<GB, 256, 0, stream>>>(x_bf, w_bf + 0 * DD, (int)N, (int)D, (int)D,
                                             MODE_SILU_STORE, nullptr, act, nullptr, nullptr);
    k_rownorm32<<<(int)(N / 32), 256, 0, stream>>>(act, (int)D, 0, k_sum, nullptr);
    // v path
    k_gemm_bf16_wmma<<<GB, 256, 0, stream>>>(x_bf, w_bf + 1 * DD, (int)N, (int)D, (int)D,
                                             MODE_SILU_COLSUM, nullptr, nullptr, nullptr, v_sum);
    // gates
    k_gemm_bf16_wmma<<<GB, 256, 0, stream>>>(x_bf, w_bf + 3 * DD, (int)N, (int)D, (int)D,
                                             MODE_SIG_COLSUM, bd, nullptr, nullptr, d_sum);
    k_gemm_bf16_wmma<<<GB, 256, 0, stream>>>(x_bf, w_bf + 4 * DD, (int)N, (int)D, (int)D,
                                             MODE_SIG_COLSUM, blr, nullptr, nullptr, lr_sum);
    k_gemm_bf16_wmma<<<GB, 256, 0, stream>>>(x_bf, w_bf + 5 * DD, (int)N, (int)D, (int)D,
                                             MODE_SIG_COLSUM, bm, nullptr, nullptr, m_sum);

    k_finalize_means<<<(int)((D + 255) / 256), 256, 0, stream>>>(
        k_sum, v_sum, d_sum, lr_sum, m_sum,
        k_mean, v_mean, alphav, thetav, etav, 1.0f / (float)N, (int)D);
    k_matvec_err<<<(int)D, 256, 0, stream>>>(state_W, k_mean, v_mean, errv, (int)D);

    // q path (act reused after k consumed it)
    k_gemm_bf16_wmma<<<GB, 256, 0, stream>>>(x_bf, w_bf + 2 * DD, (int)N, (int)D, (int)D,
                                             MODE_SILU_STORE, nullptr, act, nullptr, nullptr);
    k_rownorm32<<<(int)(N / 32), 256, 0, stream>>>(act, (int)D, 1, nullptr, q_bf);

    // W_new (bf16, slot 7)
    k_build_wnew<<<4096, 256, 0, stream>>>(state_W, state_M, alphav, thetav, etav,
                                           errv, k_mean, w_bf + 7 * DD, (int)D);

    // retrieved = q @ W_new^T
    k_gemm_bf16_wmma<<<GB, 256, 0, stream>>>(q_bf, w_bf + 7 * DD, (int)N, (int)D, (int)D,
                                             MODE_STORE_BF16, nullptr, nullptr, retr_bf, nullptr);
    // out = retrieved @ Wout^T
    k_gemm_bf16_wmma<<<GB, 256, 0, stream>>>(retr_bf, w_bf + 6 * DD, (int)N, (int)D, (int)D,
                                             MODE_STORE_F32, nullptr, (float*)d_out, nullptr, nullptr);
}
